// Attention_58428735095559
// MI455X (gfx1250) — matrix operand order repaired and checked by reading, not compiled, not run on hardware
//
#include <hip/hip_runtime.h>
#include <hip/hip_bf16.h>

#define B_ 8
#define S_ 1024
#define NH 32
#define HD 128
#define NKV 8
#define GQA 4
#define T_ (B_ * S_)
#define DKV (NKV * HD)      // 1024
#define QSTRIDE (NH * HD)   // 4096
#define NUM_SLOTS 16384

// LDS strides (shorts) with padding for conflict-free B-fragment b128 loads
#define KROW 136            // 128 + 8  (272 bytes/row)
#define VROW 72             // 64 + 8   (144 bytes/row)
#define KBUF_SHORTS (64 * KROW)
#define VBUF_SHORTS (128 * VROW)
#define KBUF_BYTES (KBUF_SHORTS * 2)   // 17408
#define VBUF_BYTES (VBUF_SHORTS * 2)   // 18432

typedef __attribute__((ext_vector_type(16))) __bf16 bf16x16;
typedef __attribute__((ext_vector_type(8)))  float  f32x8;

union FragU {
    bf16x16 v;
    unsigned u[8];
    uint4 q4[2];
};

__device__ __forceinline__ unsigned short f2bf(float f) {
    unsigned u = __float_as_uint(f);
    u += 0x7FFFu + ((u >> 16) & 1u);   // round-to-nearest-even
    return (unsigned short)(u >> 16);
}
__device__ __forceinline__ unsigned pack2(float a, float b) {
    return (unsigned)f2bf(a) | ((unsigned)f2bf(b) << 16);
}

// Kernel 1: invert slot_mapping so kv-source-token(t) = inv[seq_slot_mapping[t]].
__global__ void build_inv_kernel(const int* __restrict__ slot_mapping,
                                 int* __restrict__ inv) {
    int i = blockIdx.x * blockDim.x + threadIdx.x;
    if (i < T_) inv[slot_mapping[i]] = i;
}

// Kernel 2: one-pass gather + f32->bf16 convert.
//   kb: [b][kvh][s][d]  row-major bf16 (256B rows)
//   vt: [b][kvh][d][s]  transposed bf16 (key-contiguous)
__global__ __launch_bounds__(256)
void cvt_kernel(const float* __restrict__ k, const float* __restrict__ v,
                const int* __restrict__ seq_slot, const int* __restrict__ inv,
                unsigned short* __restrict__ kb, unsigned short* __restrict__ vt) {
    const int t   = blockIdx.x;          // token
    const int b   = t / S_;
    const int s   = t % S_;
    const int kvh = threadIdx.x >> 5;
    const int d   = (threadIdx.x & 31) * 4;
    const int j   = inv[seq_slot[t]];

    const float* kp = k + (long)j * DKV + kvh * HD + d;
    float4 k4 = *(const float4*)kp;
    uint2 pk;
    pk.x = pack2(k4.x, k4.y);
    pk.y = pack2(k4.z, k4.w);
    *(uint2*)(kb + (((long)(b * NKV + kvh)) * S_ + s) * HD + d) = pk;

    const float* vp = v + (long)j * DKV + kvh * HD + d;
    float4 v4 = *(const float4*)vp;
    const long vtb = ((long)(b * NKV + kvh)) * HD;
    vt[(vtb + d + 0) * S_ + s] = f2bf(v4.x);
    vt[(vtb + d + 1) * S_ + s] = f2bf(v4.y);
    vt[(vtb + d + 2) * S_ + s] = f2bf(v4.z);
    vt[(vtb + d + 3) * S_ + s] = f2bf(v4.w);
}

// Kernel 3: flash attention, causal, GQA; async double-buffered tile pipeline.
__global__ __launch_bounds__(128)
void flash_attn_kernel(const float* __restrict__ q,
                       const unsigned short* __restrict__ kb,
                       const unsigned short* __restrict__ vt,
                       float* __restrict__ out) {
    __shared__ __align__(16) unsigned short Kbuf[2][KBUF_SHORTS];
    __shared__ __align__(16) unsigned short Vbuf[2][VBUF_SHORTS];
    __shared__ __align__(16) unsigned short Ps[4][16 * VROW];

    const int qt = blockIdx.x;           // q tile index (64 rows)
    const int h  = blockIdx.y;
    const int bb = blockIdx.z;
    const int kh = h / GQA;

    const int tid  = threadIdx.x;
    const int wv   = tid >> 5;
    const int lane = tid & 31;
    const int nl   = lane & 15;
    const int lh   = lane >> 4;

    // ---- async-copy setup: uniform 64-bit bases + per-lane 32-bit offsets ----
    const unsigned long long baseK =
        (unsigned long long)(kb + ((long)(bb * NKV + kh)) * S_ * HD);
    const unsigned long long baseV =
        (unsigned long long)(vt + ((long)(bb * NKV + kh)) * HD * S_);
    const unsigned ksBase = (unsigned)(size_t)&Kbuf[0][0];
    const unsigned vsBase = (unsigned)(size_t)&Vbuf[0][0];
    // K tile: 64 rows x 256B;  lane li = it*128+tid -> key = li>>4, seg = tid&15
    const unsigned kLaneG   = (unsigned)((tid >> 4) * 256 + (tid & 15) * 16);
    const unsigned kLaneLds = (unsigned)((tid >> 4) * (KROW * 2) + (tid & 15) * 16);
    // V tile: 128 d-rows x 128B; lane li = it*128+tid -> d = li>>3, seg = tid&7
    const unsigned vLaneG   = (unsigned)((tid >> 3) * (S_ * 2) + (tid & 7) * 16);
    const unsigned vLaneLds = (unsigned)((tid >> 3) * (VROW * 2) + (tid & 7) * 16);

    auto issue_tile = [&](int kti) {
        const unsigned buf = (unsigned)(kti & 1);
        const unsigned gK  = (unsigned)(kti * (64 * HD * 2)) + kLaneG;     // +16KB/tile
        const unsigned lK  = ksBase + buf * KBUF_BYTES + kLaneLds;
        const unsigned gV  = (unsigned)(kti * (64 * 2)) + vLaneG;          // +128B/row
        const unsigned lV  = vsBase + buf * VBUF_BYTES + vLaneLds;
#pragma unroll
        for (int it = 0; it < 8; ++it) {
            asm volatile("global_load_async_to_lds_b128 %0, %1, %2"
                         :: "v"(lK + it * (8 * KROW * 2)), "v"(gK + it * 2048),
                            "s"(baseK) : "memory");
            asm volatile("global_load_async_to_lds_b128 %0, %1, %2"
                         :: "v"(lV + it * (16 * VROW * 2)), "v"(gV + it * 32768),
                            "s"(baseV) : "memory");
        }
    };

    // ---- Q fragments: resident in registers for whole kernel ----
    const long qrow = (long)bb * S_ + qt * 64 + wv * 16 + nl;  // token index
    FragU qf[4];
    {
        const float* qp = q + qrow * QSTRIDE + h * HD;
#pragma unroll
        for (int kk = 0; kk < 4; ++kk) {
            const float* p0 = qp + kk * 32 + lh * 8;        // A frag: K = lh*8..+7
            const float* p1 = qp + kk * 32 + 16 + lh * 8;   //         K = 16+lh*8..+7
            float4 a0 = *(const float4*)(p0);
            float4 a1 = *(const float4*)(p0 + 4);
            float4 c0 = *(const float4*)(p1);
            float4 c1 = *(const float4*)(p1 + 4);
            qf[kk].u[0] = pack2(a0.x, a0.y); qf[kk].u[1] = pack2(a0.z, a0.w);
            qf[kk].u[2] = pack2(a1.x, a1.y); qf[kk].u[3] = pack2(a1.z, a1.w);
            qf[kk].u[4] = pack2(c0.x, c0.y); qf[kk].u[5] = pack2(c0.z, c0.w);
            qf[kk].u[6] = pack2(c1.x, c1.y); qf[kk].u[7] = pack2(c1.z, c1.w);
        }
    }

    f32x8 oacc[8];
#pragma unroll
    for (int dt = 0; dt < 8; ++dt)
#pragma unroll
        for (int r = 0; r < 8; ++r) oacc[dt][r] = 0.0f;
    float mrow[8], lrow[8];
#pragma unroll
    for (int r = 0; r < 8; ++r) { mrow[r] = -3.0e38f; lrow[r] = 0.0f; }

    const float kLogScale = 0.08838834764831845f * 1.4426950408889634f; // 1/sqrt(128)*log2(e)

    issue_tile(0);  // prologue: start tile 0

    for (int kt = 0; kt <= qt; ++kt) {
        __syncthreads();      // all waves done reading buffer (kt+1)&1 from iter kt-1
        if (kt < qt) {
            issue_tile(kt + 1);
            asm volatile("s_wait_asynccnt 0x10" ::: "memory");  // tile kt complete
        } else {
            asm volatile("s_wait_asynccnt 0x0" ::: "memory");
        }
        __syncthreads();      // tile kt visible to all waves
        const unsigned short* Kc = &Kbuf[kt & 1][0];
        const unsigned short* Vc = &Vbuf[kt & 1][0];

        // ---- S = Q * K^T : 16x64 scores per wave ----
        f32x8 sf[4];
#pragma unroll
        for (int ct = 0; ct < 4; ++ct) {
            f32x8 acc;
#pragma unroll
            for (int r = 0; r < 8; ++r) acc[r] = 0.0f;
#pragma unroll
            for (int kk = 0; kk < 4; ++kk) {
                FragU bK;  // B frag: lane n = key ct*16+nl, K = lh*8..+7, 16+lh*8..+7
                const unsigned short* base = Kc + (ct * 16 + nl) * KROW + kk * 32 + lh * 8;
                bK.q4[0] = *(const uint4*)(base);
                bK.q4[1] = *(const uint4*)(base + 16);
                acc = __builtin_amdgcn_wmma_f32_16x16x32_bf16(
                    false, qf[kk].v, false, bK.v, (short)0, acc, false, false);
            }
            sf[ct] = acc;
        }
        // scale (with log2e folded) + causal mask on diagonal tile
#pragma unroll
        for (int ct = 0; ct < 4; ++ct)
#pragma unroll
            for (int r = 0; r < 8; ++r) sf[ct][r] *= kLogScale;
        if (kt == qt) {
#pragma unroll
            for (int ct = 0; ct < 4; ++ct) {
                const int ng = ct * 16 + nl;              // key within tile
#pragma unroll
                for (int r = 0; r < 8; ++r) {
                    const int mg = wv * 16 + r + 8 * lh;  // q row within tile
                    sf[ct][r] = (ng > mg) ? -3.0e38f : sf[ct][r];
                }
            }
        }

        // ---- online softmax (rows live on 16-lane halves) ----
        float alpha[8];
#pragma unroll
        for (int r = 0; r < 8; ++r) {
            float m0 = fmaxf(fmaxf(sf[0][r], sf[1][r]), fmaxf(sf[2][r], sf[3][r]));
#pragma unroll
            for (int off = 1; off < 16; off <<= 1)
                m0 = fmaxf(m0, __shfl_xor(m0, off, 32));
            const float mn = fmaxf(mrow[r], m0);
            alpha[r] = exp2f(mrow[r] - mn);
            mrow[r]  = mn;
        }
#pragma unroll
        for (int r = 0; r < 8; ++r) {
            float s0 = 0.0f;
#pragma unroll
            for (int ct = 0; ct < 4; ++ct) {
                const float p = exp2f(sf[ct][r] - mrow[r]);
                sf[ct][r] = p;
                s0 += p;
            }
#pragma unroll
            for (int off = 1; off < 16; off <<= 1)
                s0 += __shfl_xor(s0, off, 32);
            lrow[r] = lrow[r] * alpha[r] + s0;
        }
#pragma unroll
        for (int dt = 0; dt < 8; ++dt)
#pragma unroll
            for (int r = 0; r < 8; ++r) oacc[dt][r] *= alpha[r];

        // ---- stage P (bf16) through per-wave LDS to reach A-fragment layout ----
#pragma unroll
        for (int ct = 0; ct < 4; ++ct)
#pragma unroll
            for (int r = 0; r < 8; ++r)
                Ps[wv][(r + 8 * lh) * VROW + ct * 16 + nl] = f2bf(sf[ct][r]);
        asm volatile("s_wait_dscnt 0" ::: "memory");  // same-wave cross-lane LDS RAW

        FragU pf[2];
#pragma unroll
        for (int kk = 0; kk < 2; ++kk) {
            const unsigned short* base = &Ps[wv][nl * VROW + kk * 32 + lh * 8];
            pf[kk].q4[0] = *(const uint4*)(base);
            pf[kk].q4[1] = *(const uint4*)(base + 16);
        }
        // ---- O += P * V ----
#pragma unroll
        for (int dt = 0; dt < 8; ++dt) {
#pragma unroll
            for (int kk = 0; kk < 2; ++kk) {
                FragU bV;  // lane n = out dim dt*16+nl, keys lh*8..+7, 16+lh*8..+7
                const unsigned short* base = Vc + (dt * 16 + nl) * VROW + kk * 32 + lh * 8;
                bV.q4[0] = *(const uint4*)(base);
                bV.q4[1] = *(const uint4*)(base + 16);
                oacc[dt] = __builtin_amdgcn_wmma_f32_16x16x32_bf16(
                    false, pf[kk].v, false, bV.v, (short)0, oacc[dt], false, false);
            }
        }
    }

    // ---- finalize: O /= l, write f32 output ----
#pragma unroll
    for (int r = 0; r < 8; ++r) {
        const float rl = 1.0f / lrow[r];
        const long row = (long)bb * S_ + qt * 64 + wv * 16 + r + 8 * lh;
        float* op = out + row * QSTRIDE + h * HD + nl;
#pragma unroll
        for (int dt = 0; dt < 8; ++dt)
            op[dt * 16] = oacc[dt][r] * rl;
    }
}

extern "C" void kernel_launch(void* const* d_in, const int* in_sizes, int n_in,
                              void* d_out, int out_size, void* d_ws, size_t ws_size,
                              hipStream_t stream) {
    (void)in_sizes; (void)n_in; (void)out_size; (void)ws_size;
    const float* q        = (const float*)d_in[0];
    const float* k        = (const float*)d_in[1];
    const float* v        = (const float*)d_in[2];
    const int*   slot_map = (const int*)d_in[5];
    const int*   seq_slot = (const int*)d_in[6];
    float*       out      = (float*)d_out;

    // workspace layout: inv (64KB) | kb bf16 (16.78MB) | vt bf16 (16.78MB)
    char* ws = (char*)d_ws;
    int*            inv = (int*)ws;
    unsigned short* kb  = (unsigned short*)(ws + (1 << 16));
    unsigned short* vt  = (unsigned short*)(ws + (1 << 16) + (size_t)T_ * DKV * 2);

    build_inv_kernel<<<(T_ + 255) / 256, 256, 0, stream>>>(slot_map, inv);
    cvt_kernel<<<T_, 256, 0, stream>>>(k, v, seq_slot, inv, kb, vt);
    dim3 grid(S_ / 64, NH, B_);
    flash_attn_kernel<<<grid, dim3(128), 0, stream>>>(q, kb, vt, out);
}
